// FusionCrossAttention_30442728194571
// MI455X (gfx1250) — matrix operand order repaired and checked by reading, not compiled, not run on hardware
//
#include <hip/hip_runtime.h>

#define USE_ASYNC_LDS 1   // gfx1250 GLOBAL_LOAD_ASYNC_TO_LDS_* path (set 0 for sync fallback)

typedef __bf16 bf16;
typedef __attribute__((ext_vector_type(4)))  __bf16 v4bf;
typedef __attribute__((ext_vector_type(8)))  __bf16 v8bf;
typedef __attribute__((ext_vector_type(16))) __bf16 v16bf;
typedef __attribute__((ext_vector_type(8)))  float  v8f;

constexpr int DF = 1024;    // fusion dim
constexpr int BR = 16384;   // batch rows

// ---------------------------------------------------------------------------
// CDNA5 async global->LDS copy (16B per lane), tracked by ASYNCcnt.
// lds_addr = low 32 bits of generic shared pointer (ISA: flat->LDS truncation).
// ---------------------------------------------------------------------------
__device__ __forceinline__ void async_copy16(unsigned lds_addr, const bf16* g)
{
#if USE_ASYNC_LDS
    asm volatile("global_load_async_to_lds_b128 %0, %1, off"
                 :: "v"(lds_addr), "v"(g)
                 : "memory");
#else
    (void)lds_addr; (void)g;
#endif
}

__device__ __forceinline__ void wait_async0()
{
#if __has_builtin(__builtin_amdgcn_s_wait_asynccnt)
    __builtin_amdgcn_s_wait_asynccnt(0);
#else
    asm volatile("s_wait_asynccnt 0" ::: "memory");
#endif
}

// ---------------------------------------------------------------------------
// Generic bf16 WMMA GEMM: C[m,n] = sum_k A[m,k] * Bt[k,n]  (+bias +res, relu)
// A: [M,K] bf16 row-major.  Bt: [K,N] bf16 row-major (i.e. W^T, K-major).
// Tile: 128x128, BK=32 (one v_wmma_f32_16x16x32_bf16 k-step per tile),
// 8 waves (2x4), each wave owns 64x32 = 4x2 WMMA fragments.
// Epilogue is compile-time specialized -> branch-free.
// ---------------------------------------------------------------------------
template<bool BIAS, bool RES, bool RELU, bool OUTF, bool OUTB>
__global__ __launch_bounds__(256)
void wmma_gemm(const bf16* __restrict__ A, const bf16* __restrict__ Bt,
               const float* __restrict__ bias, const float* __restrict__ res,
               float* __restrict__ outF, bf16* __restrict__ outB,
               int M, int N, int K)
{
    constexpr int SA = 40;   // LDS A row stride (halves): 32 data + pad, 16B aligned
    constexpr int SB = 136;  // LDS B row stride (halves): 128 data + pad, 16B aligned
    __shared__ __align__(16) bf16 sA[2][128 * SA];
    __shared__ __align__(16) bf16 sB[2][32 * SB];

    const int tid   = threadIdx.x;
    const int lane  = tid & 31;
    const int wave  = tid >> 5;
    const int waveM = wave >> 2;   // 0..1
    const int waveN = wave & 3;    // 0..3
    const int bm = blockIdx.y * 128;
    const int bn = blockIdx.x * 128;

    // global->LDS copy mapping: 16B chunks, 2 per thread per (A,B) tile
    const int ar = tid >> 2;            // A rows ar and ar+64
    const int ao = (tid & 3) * 8;       // half offset within A row
    const int br = tid >> 4;            // B k-rows br and br+16
    const int bo = (tid & 15) * 8;      // half offset within B row

    const bf16* gA0 = A  + (size_t)(bm + ar) * K + ao;
    const bf16* gA1 = gA0 + (size_t)64 * K;
    const bf16* gB0 = Bt + (size_t)br * N + bn + bo;
    const bf16* gB1 = gB0 + (size_t)16 * N;

    v8f acc[4][2];
    #pragma unroll
    for (int i = 0; i < 4; ++i)
        #pragma unroll
        for (int j = 0; j < 2; ++j)
            #pragma unroll
            for (int e = 0; e < 8; ++e) acc[i][j][e] = 0.0f;

    const int nt = K >> 5;

#if USE_ASYNC_LDS
    const unsigned lA0 = (unsigned)(size_t)&sA[0][ar * SA + ao];
    const unsigned lA1 = (unsigned)(size_t)&sA[0][(ar + 64) * SA + ao];
    const unsigned lB0 = (unsigned)(size_t)&sB[0][br * SB + bo];
    const unsigned lB1 = (unsigned)(size_t)&sB[0][(br + 16) * SB + bo];
    constexpr unsigned APITCH = 128 * SA * 2;   // bytes per A buffer
    constexpr unsigned BPITCH = 32 * SB * 2;    // bytes per B buffer

    auto issue_tile = [&](int buf, int kt) {
        async_copy16(lA0 + buf * APITCH, gA0 + (size_t)kt * 32);
        async_copy16(lA1 + buf * APITCH, gA1 + (size_t)kt * 32);
        async_copy16(lB0 + buf * BPITCH, gB0 + (size_t)kt * 32 * N);
        async_copy16(lB1 + buf * BPITCH, gB1 + (size_t)kt * 32 * N);
    };
    issue_tile(0, 0);
    wait_async0();
    __syncthreads();
#else
    uint4 ra0 = *(const uint4*)gA0;
    uint4 ra1 = *(const uint4*)gA1;
    uint4 rb0 = *(const uint4*)gB0;
    uint4 rb1 = *(const uint4*)gB1;
    *(uint4*)&sA[0][ar * SA + ao]        = ra0;
    *(uint4*)&sA[0][(ar + 64) * SA + ao] = ra1;
    *(uint4*)&sB[0][br * SB + bo]        = rb0;
    *(uint4*)&sB[0][(br + 16) * SB + bo] = rb1;
    __syncthreads();
#endif

    const int mrow  = lane & 15;          // A frag row within 16-tile
    const int khalf = (lane >> 4) * 8;    // A frag k offset (halves)

    for (int kt = 0; kt < nt; ++kt) {
        const int cur   = kt & 1;
        const bool more = (kt + 1) < nt;

#if USE_ASYNC_LDS
        if (more) issue_tile(cur ^ 1, kt + 1);   // fill runs under the WMMA stream
#else
        if (more) {
            ra0 = *(const uint4*)(gA0 + (size_t)(kt + 1) * 32);
            ra1 = *(const uint4*)(gA1 + (size_t)(kt + 1) * 32);
            rb0 = *(const uint4*)(gB0 + (size_t)(kt + 1) * 32 * N);
            rb1 = *(const uint4*)(gB1 + (size_t)(kt + 1) * 32 * N);
        }
#endif
        if (kt + 2 < nt) {
            __builtin_prefetch(gA0 + (size_t)(kt + 2) * 32, 0, 1);
            __builtin_prefetch(gB0 + (size_t)(kt + 2) * 32 * N, 0, 1);
        }

        const bf16* sa = &sA[cur][0];
        const bf16* sb = &sB[cur][0];

        // A fragments: lane<16 -> k 0..7 & 16..23 ; lane>=16 -> k 8..15 & 24..31
        v16bf af[4];
        #pragma unroll
        for (int i = 0; i < 4; ++i) {
            const bf16* p = sa + (waveM * 64 + i * 16 + mrow) * SA + khalf;
            v8bf lo = *(const v8bf*)p;
            v8bf hi = *(const v8bf*)(p + 16);
            af[i] = __builtin_shufflevector(lo, hi, 0,1,2,3,4,5,6,7,8,9,10,11,12,13,14,15);
        }
        // B fragments: lane holds column n = lane&15; lane<16 -> k 0..7 & 16..23 ; lane>=16 -> k 8..15 & 24..31
        v16bf bfv[2];
        #pragma unroll
        for (int j = 0; j < 2; ++j) {
            const bf16* p = sb + khalf * SB + waveN * 32 + j * 16 + mrow;
            #pragma unroll
            for (int s = 0; s < 8; ++s) {
                bfv[j][s]     = p[s * SB];
                bfv[j][8 + s] = p[(16 + s) * SB];
            }
        }

        #pragma unroll
        for (int i = 0; i < 4; ++i)
            #pragma unroll
            for (int j = 0; j < 2; ++j)
                acc[i][j] = __builtin_amdgcn_wmma_f32_16x16x32_bf16(
                    false, af[i], false, bfv[j], (short)0, acc[i][j], false, false);

        if (more) {
#if USE_ASYNC_LDS
            wait_async0();        // next buffer fully landed in LDS
            __syncthreads();
#else
            const int nb = cur ^ 1;
            *(uint4*)&sA[nb][ar * SA + ao]        = ra0;
            *(uint4*)&sA[nb][(ar + 64) * SA + ao] = ra1;
            *(uint4*)&sB[nb][br * SB + bo]        = rb0;
            *(uint4*)&sB[nb][(br + 16) * SB + bo] = rb1;
            __syncthreads();
#endif
        }
    }

    // epilogue: C layout -> VGPR e, lane l: m = e + 8*(l>>4), n = l&15
    const int mh = (lane >> 4) * 8;
    const int nl = lane & 15;
    #pragma unroll
    for (int i = 0; i < 4; ++i) {
        #pragma unroll
        for (int j = 0; j < 2; ++j) {
            const int n = bn + waveN * 32 + j * 16 + nl;
            float bv = 0.0f;
            if constexpr (BIAS) bv = bias[n];
            #pragma unroll
            for (int e = 0; e < 8; ++e) {
                const int m = bm + waveM * 64 + i * 16 + mh + e;
                float v = acc[i][j][e] + bv;
                if constexpr (RES)  v += res[(size_t)m * N + n];
                if constexpr (RELU) v = fmaxf(v, 0.0f);
                if constexpr (OUTF) outF[(size_t)m * N + n] = v;
                if constexpr (OUTB) outB[(size_t)m * N + n] = (bf16)v;
            }
        }
    }
}

// ---------------------------------------------------------------------------
// LayerNorm over rows of 1024: y = (x-mu)*rsqrt(var+eps)*g + b
// ---------------------------------------------------------------------------
__global__ __launch_bounds__(128)
void ln_kernel(const float* __restrict__ x, const float* __restrict__ g,
               const float* __restrict__ b, float* __restrict__ outF,
               bf16* __restrict__ outB)
{
    const int row = blockIdx.x;
    const int tid = threadIdx.x;
    const float* xr = x + (size_t)row * DF;
    float v[8];
    float s = 0.0f, sq = 0.0f;
    #pragma unroll
    for (int e = 0; e < 8; ++e) {
        v[e] = xr[tid + e * 128];
        s += v[e]; sq += v[e] * v[e];
    }
    #pragma unroll
    for (int o = 16; o > 0; o >>= 1) { s += __shfl_xor(s, o); sq += __shfl_xor(sq, o); }
    __shared__ float red[8];
    const int wv = tid >> 5, ln = tid & 31;
    if (ln == 0) { red[wv] = s; red[4 + wv] = sq; }
    __syncthreads();
    s  = red[0] + red[1] + red[2] + red[3];
    sq = red[4] + red[5] + red[6] + red[7];
    const float mu  = s * (1.0f / DF);
    const float var = sq * (1.0f / DF) - mu * mu;
    const float rs  = rsqrtf(var + 1e-5f);
    #pragma unroll
    for (int e = 0; e < 8; ++e) {
        const int n = tid + e * 128;
        const float y = (v[e] - mu) * rs * g[n] + b[n];
        if (outF) outF[(size_t)row * DF + n] = y;
        if (outB) outB[(size_t)row * DF + n] = (bf16)y;
    }
}

// src f32 [R,C] -> dst bf16 [C,R]  (K-major B-operand prep), via LDS tile
__global__ __launch_bounds__(256)
void transpose_cvt(const float* __restrict__ src, bf16* __restrict__ dst, int R, int C)
{
    __shared__ float t[32][33];
    const int tx = threadIdx.x, ty = threadIdx.y;
    const int c0 = blockIdx.x * 32, r0 = blockIdx.y * 32;
    #pragma unroll
    for (int i = 0; i < 4; ++i)
        t[ty + 8 * i][tx] = src[(size_t)(r0 + ty + 8 * i) * C + c0 + tx];
    __syncthreads();
    #pragma unroll
    for (int i = 0; i < 4; ++i)
        dst[(size_t)(c0 + ty + 8 * i) * R + r0 + tx] = (bf16)t[tx][ty + 8 * i];
}

// plain f32 -> bf16 convert (A-operand prep)
__global__ __launch_bounds__(256)
void cvt_bf16(const float* __restrict__ src, bf16* __restrict__ dst, int n)
{
    const int i = (blockIdx.x * 256 + threadIdx.x) * 4;
    if (i + 3 < n) {
        const float4 f = *(const float4*)(src + i);
        v4bf o;
        o[0] = (bf16)f.x; o[1] = (bf16)f.y; o[2] = (bf16)f.z; o[3] = (bf16)f.w;
        *(v4bf*)(dst + i) = o;
    }
}

// b'[blk][n] = out_w[blk] @ bv[blk] + out_b[blk]   (one wave per output)
__global__ __launch_bounds__(256)
void bias_fuse(const float* __restrict__ out_w, const float* __restrict__ in_b,
               const float* __restrict__ out_b, float* __restrict__ bf)
{
    const int gw = (blockIdx.x * 256 + threadIdx.x) >> 5;
    const int ln = threadIdx.x & 31;
    const int blk = gw >> 10, n = gw & 1023;
    const float* w  = out_w + ((size_t)blk * DF + n) * DF;
    const float* bv = in_b + (size_t)blk * 3 * DF + 2 * DF;
    float s = 0.0f;
    for (int m = ln; m < DF; m += 32) s += w[m] * bv[m];
    #pragma unroll
    for (int o = 16; o > 0; o >>= 1) s += __shfl_xor(s, o);
    if (ln == 0) bf[(size_t)blk * DF + n] = s + out_b[(size_t)blk * DF + n];
}

// out[:, pair*D:(pair+1)*D] = (oe + oo) * 0.5 * gate[pair]
__global__ __launch_bounds__(256)
void combine_out(const float* __restrict__ oe, const float* __restrict__ oo,
                 const float* __restrict__ gate, int pair, float* __restrict__ out)
{
    const size_t idx = (size_t)blockIdx.x * 256 + threadIdx.x;
    const size_t i = idx >> 10, n = idx & 1023;
    out[i * (3 * DF) + (size_t)pair * DF + n] = (oe[idx] + oo[idx]) * 0.5f * gate[pair];
}

// ---------------------------------------------------------------------------
extern "C" void kernel_launch(void* const* d_in, const int* in_sizes, int n_in,
                              void* d_out, int out_size, void* d_ws, size_t ws_size,
                              hipStream_t stream)
{
    (void)in_sizes; (void)n_in; (void)out_size; (void)ws_size;
    const float* invec[3] = {(const float*)d_in[0], (const float*)d_in[1], (const float*)d_in[2]};
    const float* pw[3]    = {(const float*)d_in[3], (const float*)d_in[5], (const float*)d_in[7]};
    const float* pb[3]    = {(const float*)d_in[4], (const float*)d_in[6], (const float*)d_in[8]};
    const float* in_w  = (const float*)d_in[9];
    const float* in_b  = (const float*)d_in[10];
    const float* out_w = (const float*)d_in[11];
    const float* out_b = (const float*)d_in[12];
    const float* ff1_w = (const float*)d_in[13];
    const float* ff1_b = (const float*)d_in[14];
    const float* ff2_w = (const float*)d_in[15];
    const float* ff2_b = (const float*)d_in[16];
    const float* ln1_g = (const float*)d_in[17];
    const float* ln1_b = (const float*)d_in[18];
    const float* ln2_g = (const float*)d_in[19];
    const float* ln2_b = (const float*)d_in[20];
    const float* gate  = (const float*)d_in[21];
    float* out = (float*)d_out;

    // bump allocator over workspace
    char* base = (char*)d_ws;
    size_t off = 0;
    auto alloc = [&](size_t bytes) -> void* {
        void* p = base + off;
        off += bytes; off = (off + 255) & ~(size_t)255;
        return p;
    };

    bf16* projT[3]; for (int i = 0; i < 3; ++i) projT[i] = (bf16*)alloc(2ull * DF * DF);
    bf16* wvT[6];   for (int i = 0; i < 6; ++i) wvT[i]   = (bf16*)alloc(2ull * DF * DF);
    bf16* owT[6];   for (int i = 0; i < 6; ++i) owT[i]   = (bf16*)alloc(2ull * DF * DF);
    bf16* f1T[6];   for (int i = 0; i < 6; ++i) f1T[i]   = (bf16*)alloc(2ull * DF * 2 * DF);
    bf16* f2T[6];   for (int i = 0; i < 6; ++i) f2T[i]   = (bf16*)alloc(2ull * DF * 2 * DF);
    bf16* wfus[6];  for (int i = 0; i < 6; ++i) wfus[i]  = (bf16*)alloc(2ull * DF * DF);
    float* bfus = (float*)alloc(6ull * DF * 4);
    bf16* inb[3];  for (int i = 0; i < 3; ++i) inb[i] = (bf16*)alloc(2ull * BR * DF);
    float* qf[3];  for (int i = 0; i < 3; ++i) qf[i]  = (float*)alloc(4ull * BR * DF);
    bf16* qb[3];   for (int i = 0; i < 3; ++i) qb[i]  = (bf16*)alloc(2ull * BR * DF);
    float* sbuf = (float*)alloc(4ull * BR * DF);
    float* xf   = (float*)alloc(4ull * BR * DF);
    bf16*  xb   = (bf16*)alloc(2ull * BR * DF);
    bf16*  hb   = (bf16*)alloc(2ull * BR * 2 * DF);
    float* oe   = (float*)alloc(4ull * BR * DF);
    float* oo   = (float*)alloc(4ull * BR * DF);

    const dim3 tb(32, 8);
    // --- weight prep: transpose-convert everything to K-major bf16 ---
    for (int i = 0; i < 3; ++i)
        transpose_cvt<<<dim3(DF / 32, DF / 32), tb, 0, stream>>>(pw[i], projT[i], DF, DF);
    for (int b = 0; b < 6; ++b) {
        transpose_cvt<<<dim3(DF / 32, DF / 32), tb, 0, stream>>>(
            in_w + ((size_t)b * 3 * DF + 2 * DF) * DF, wvT[b], DF, DF);   // wv slice
        transpose_cvt<<<dim3(DF / 32, DF / 32), tb, 0, stream>>>(
            out_w + (size_t)b * DF * DF, owT[b], DF, DF);
        transpose_cvt<<<dim3(DF / 32, (2 * DF) / 32), tb, 0, stream>>>(
            ff1_w + (size_t)b * 2 * DF * DF, f1T[b], 2 * DF, DF);
        transpose_cvt<<<dim3((2 * DF) / 32, DF / 32), tb, 0, stream>>>(
            ff2_w + (size_t)b * DF * 2 * DF, f2T[b], DF, 2 * DF);
    }
    bias_fuse<<<768, 256, 0, stream>>>(out_w, in_b, out_b, bfus);

    // --- fused V/out weight: Wfus[k,n] = sum_m wv[m,k] * out_w[n,m] ---
    for (int b = 0; b < 6; ++b)
        wmma_gemm<false, false, false, false, true>
            <<<dim3(DF / 128, DF / 128), 256, 0, stream>>>(
                wvT[b], owT[b], nullptr, nullptr, nullptr, wfus[b], DF, DF, DF);

    // --- input projections ---
    for (int i = 0; i < 3; ++i) {
        cvt_bf16<<<(BR * DF / 4) / 256, 256, 0, stream>>>(invec[i], inb[i], BR * DF);
        wmma_gemm<true, false, false, true, true>
            <<<dim3(DF / 128, BR / 128), 256, 0, stream>>>(
                inb[i], projT[i], pb[i], nullptr, qf[i], qb[i], BR, DF, DF);
    }

    // --- 6 fusion blocks (softmax over singleton == 1 -> q/k projections elided) ---
    const int qidx[6] = {0, 0, 1, 1, 2, 2};
    const int kidx[6] = {1, 2, 0, 2, 0, 1};
    for (int b = 0; b < 6; ++b) {
        // attn = kv @ Wfus + b'  (+ residual q)
        wmma_gemm<true, true, false, true, false>
            <<<dim3(DF / 128, BR / 128), 256, 0, stream>>>(
                qb[kidx[b]], wfus[b], bfus + (size_t)b * DF, qf[qidx[b]], sbuf, nullptr,
                BR, DF, DF);
        ln_kernel<<<BR, 128, 0, stream>>>(sbuf, ln1_g + (size_t)b * DF, ln1_b + (size_t)b * DF, xf, xb);
        // h = relu(x @ ff1^T + b1)
        wmma_gemm<true, false, true, false, true>
            <<<dim3((2 * DF) / 128, BR / 128), 256, 0, stream>>>(
                xb, f1T[b], ff1_b + (size_t)b * 2 * DF, nullptr, nullptr, hb, BR, 2 * DF, DF);
        // s = h @ ff2^T + b2 + x
        wmma_gemm<true, true, false, true, false>
            <<<dim3(DF / 128, BR / 128), 256, 0, stream>>>(
                hb, f2T[b], ff2_b + (size_t)b * DF, xf, sbuf, nullptr, BR, DF, 2 * DF);
        ln_kernel<<<BR, 128, 0, stream>>>(sbuf, ln2_g + (size_t)b * DF, ln2_b + (size_t)b * DF,
                                          (b & 1) ? oo : oe, (bf16*)nullptr);
        if (b & 1)
            combine_out<<<(BR * DF) / 256, 256, 0, stream>>>(oe, oo, gate, b >> 1, out);
    }
}
